// MultiHeadedAttention_31954556682880
// MI455X (gfx1250) — compile-verified, matrix operand order checked by reading, not run on hardware
//
#include <hip/hip_runtime.h>

// ---------------- problem constants ----------------
#define BB 4
#define SS 2048
#define DD 512
#define HH 8
#define DKK 64
#define MM (BB * SS)   // 8192
#define NN DD          // 512
#define KK DD          // 512

// ---------------- vector types ----------------
typedef __attribute__((ext_vector_type(16))) __bf16 v16bf;
typedef __attribute__((ext_vector_type(8)))  float  v8f;
typedef __attribute__((ext_vector_type(4)))  float  v4f;
typedef __attribute__((ext_vector_type(8)))  unsigned short v8u;
typedef __attribute__((ext_vector_type(16))) unsigned short v16u;
typedef __attribute__((ext_vector_type(4)))  unsigned int u32x4;
typedef __attribute__((ext_vector_type(8)))  unsigned int u32x8;

// ---------------- helpers ----------------
__device__ __forceinline__ unsigned short f2bf(float f) {
  unsigned int u = __float_as_uint(f);
  unsigned int r = u + 0x7fffu + ((u >> 16) & 1u);   // RNE
  return (unsigned short)(r >> 16);
}
__device__ __forceinline__ float bf2f(unsigned short u) {
  return __uint_as_float(((unsigned int)u) << 16);
}

// Load one 16x16x32 bf16 WMMA A/B fragment (16 bf16 per lane).
// Lane half h=lane/16 selects K ranges [8h,8h+8) and [16+8h,24+8h).
__device__ __forceinline__ v16u load_frag(const unsigned short* base, int h) {
  v8u lo = *(const v8u*)(base + 8 * h);
  v8u hi = *(const v8u*)(base + 16 + 8 * h);
  v16u r;
#pragma unroll
  for (int i = 0; i < 8; ++i) { r[i] = lo[i]; r[8 + i] = hi[i]; }
  return r;
}

__device__ __forceinline__ v8f wmma_bf16(v16u a, v16u b, v8f c) {
  return __builtin_amdgcn_wmma_f32_16x16x32_bf16(
      false, __builtin_bit_cast(v16bf, a),
      false, __builtin_bit_cast(v16bf, b),
      (short)0, c, false, false);
}

// ---------------- Tensor Data Mover: 2-D tile load, bf16 elements ----------------
// Loads a 64x64 tile of 2-byte elements (row stride `stride_elems` in memory)
// into LDS at `lds_addr`, with 16B padding inserted after every 128B row
// (LDS row pitch becomes 144B = 72 bf16, 16B aligned -> conflict-free b128 reads).
// D# built per CDNA5 ISA 8.3/8.4; issued via VIMAGE TENSOR_LOAD_TO_LDS.
__device__ __forceinline__ void tdm_load_tile64(unsigned lds_addr, const void* gaddr,
                                                unsigned stride_elems) {
  unsigned long long ga = (unsigned long long)(uintptr_t)gaddr;
  u32x4 g0;
  g0[0] = 1u;                                   // count=1, user descriptor
  g0[1] = lds_addr;                             // lds_addr [63:32]
  g0[2] = (unsigned)ga;                         // global_addr low
  g0[3] = (unsigned)(ga >> 32) | 0x80000000u;   // global_addr hi | type=2 @ [127:126]
  u32x8 g1;
  g1[0] = 0x10000u                              // data_size=1 (2 bytes)
        | (1u << 20)                            // pad_enable
        | (4u << 22)                            // pad_interval: 32 DWORDs (128B)
        | (3u << 25);                           // pad_amount: 4 DWORDs (16B)
  g1[1] = 64u << 16;                            // tensor_dim0 = 64 (low16 @ [63:48])
  g1[2] = 64u << 16;                            // tensor_dim1 = 64 (low16 @ [95:80] of field [111:80])
  g1[3] = 64u << 16;                            // tile_dim0 = 64 @ [127:112]
  g1[4] = 64u;                                  // tile_dim1 = 64 @ [143:128]; tile_dim2=0
  g1[5] = stride_elems;                         // tensor_dim0_stride low32 @ [191:160]
  g1[6] = 0u;                                   // stride hi / dim1_stride lo
  g1[7] = 0u;
  asm volatile("tensor_load_to_lds %0, %1" :: "s"(g0), "s"(g1) : "memory");
}

// =====================================================================
// GEMM: Y = X(f32)[M,K] @ W(f32)[K,N] + bias   (mode compile-time)
// MODE 0: f32 flat [M,N]    MODE 1: bf16 [B,H,S,DK]    MODE 2: bf16 [B,H,DK,S]
// =====================================================================
template <int MODE>
__global__ __launch_bounds__(256) void gemm_bf16_kernel(
    const float* __restrict__ A, const float* __restrict__ W,
    const float* __restrict__ bias, float* __restrict__ outF,
    unsigned short* __restrict__ outB) {
  __shared__ __align__(16) unsigned short As[64][40];  // 80B rows
  __shared__ __align__(16) unsigned short Wt[64][40];  // Wt[n][k]

  const int tid  = threadIdx.x;
  const int wave = tid >> 5, lane = tid & 31;
  const int l16  = lane & 15, lh = lane >> 4;
  const int m0 = blockIdx.y * 64, n0 = blockIdx.x * 64;
  const int wm = wave >> 1, wn = wave & 1;   // 4x2 wave grid of 16x32 sub-tiles

  v8f acc0 = {0.f,0.f,0.f,0.f,0.f,0.f,0.f,0.f};
  v8f acc1 = {0.f,0.f,0.f,0.f,0.f,0.f,0.f,0.f};

  for (int k0 = 0; k0 < KK; k0 += 32) {
    { // A tile 64x32 f32 -> bf16 LDS
      int r  = tid >> 2;
      int c0 = (tid & 3) * 8;
      const float* src = A + (size_t)(m0 + r) * KK + k0 + c0;
      v4f f0 = *(const v4f*)(src);
      v4f f1 = *(const v4f*)(src + 4);
      v8u p;
#pragma unroll
      for (int i = 0; i < 4; ++i) { p[i] = f2bf(f0[i]); p[4 + i] = f2bf(f1[i]); }
      *(v8u*)&As[r][c0] = p;
    }
    { // W tile 32x64 f32 -> transposed bf16 LDS Wt[n][k]
#pragma unroll
      for (int i = 0; i < 8; ++i) {
        int e  = i * 256 + tid;
        int kk = e >> 6;
        int n  = e & 63;
        Wt[n][kk] = f2bf(W[(size_t)(k0 + kk) * NN + n0 + n]);
      }
    }
    __syncthreads();

    v16u afrag = load_frag(&As[wm * 16 + l16][0], lh);
    int bn = wn * 32 + l16;
    v16u b0 = load_frag(&Wt[bn][0], lh);
    v16u b1 = load_frag(&Wt[bn + 16][0], lh);
    acc0 = wmma_bf16(afrag, b0, acc0);
    acc1 = wmma_bf16(afrag, b1, acc1);
    __syncthreads();
  }

  // epilogue: C layout rows m = r + 8*lh, col n = lane%16 (+subtile)
#pragma unroll
  for (int st = 0; st < 2; ++st) {
    v8f accv = st ? acc1 : acc0;
    int n = n0 + wn * 32 + st * 16 + l16;
    float bv = bias[n];
#pragma unroll
    for (int r = 0; r < 8; ++r) {
      int m = m0 + wm * 16 + r + 8 * lh;
      float val = accv[r] + bv;
      if (MODE == 0) {
        outF[(size_t)m * NN + n] = val;
      } else {
        int b  = m >> 11;      // m / S
        int s  = m & (SS - 1);
        int hd = n >> 6;       // n / DK
        int dk = n & (DKK - 1);
        if (MODE == 1)
          outB[(((size_t)(b * HH + hd)) * SS + s) * DKK + dk] = f2bf(val);
        else
          outB[(((size_t)(b * HH + hd)) * DKK + dk) * SS + s] = f2bf(val);
      }
    }
  }
}

// =====================================================================
// lex pipeline
// =====================================================================
__global__ __launch_bounds__(256) void lex_matvec_kernel(
    const float* __restrict__ Lex, const float* __restrict__ Wl,
    const float* __restrict__ bl, float* __restrict__ lexrow) {
  int j = blockIdx.x * 256 + threadIdx.x;
  int b = blockIdx.y;
  const float* lp = Lex + (size_t)b * SS;
  float acc = bl[j];
  for (int i = 0; i < SS; ++i)
    acc += (lp[i] * 0.001f) * Wl[(size_t)i * SS + j];
  lexrow[(size_t)b * SS + j] = acc;
}

__global__ __launch_bounds__(256) void lex_softmax_kernel(
    const float* __restrict__ lexrow, const int* __restrict__ mask,
    float* __restrict__ lexattn) {
  __shared__ float redA[8], redB[8];
  int b = blockIdx.x, tid = threadIdx.x;
  float v[8];
  float mx = -3.0e38f;
#pragma unroll
  for (int i = 0; i < 8; ++i) {
    int j = tid + i * 256;
    float x = (mask[(size_t)b * SS + j] == 1) ? -1e9f : lexrow[(size_t)b * SS + j];
    v[i] = x;
    mx = fmaxf(mx, x);
  }
  for (int o = 16; o > 0; o >>= 1) mx = fmaxf(mx, __shfl_xor(mx, o, 32));
  if ((tid & 31) == 0) redA[tid >> 5] = mx;
  __syncthreads();
  float bm = redA[0];
#pragma unroll
  for (int w = 1; w < 8; ++w) bm = fmaxf(bm, redA[w]);
  float s = 0.f;
#pragma unroll
  for (int i = 0; i < 8; ++i) { v[i] = __expf(v[i] - bm); s += v[i]; }
  for (int o = 16; o > 0; o >>= 1) s += __shfl_xor(s, o, 32);
  if ((tid & 31) == 0) redB[tid >> 5] = s;
  __syncthreads();
  float bs = 0.f;
#pragma unroll
  for (int w = 0; w < 8; ++w) bs += redB[w];
  float inv = 1.0f / bs;
#pragma unroll
  for (int i = 0; i < 8; ++i)
    lexattn[(size_t)b * SS + tid + i * 256] = v[i] * inv;
}

__global__ __launch_bounds__(64) void lv_kernel(
    const float* __restrict__ lexattn, const unsigned short* __restrict__ vT,
    float* __restrict__ lv) {
  int bh = blockIdx.x;           // b*H + h
  int dk = threadIdx.x;
  int b  = bh / HH;
  const unsigned short* vp = vT + ((size_t)bh * DKK + dk) * SS;
  const float* la = lexattn + (size_t)b * SS;
  float acc = 0.f;
  for (int s = 0; s < SS; ++s) acc += la[s] * bf2f(vp[s]);
  lv[(size_t)bh * DKK + dk] = acc;
}

// =====================================================================
// Flash attention with TDM double-buffered K/V staging.
// out = 0.5 * softmax(QK^T/8 masked) @ V + 0.5 * lv   -> concat [B,S,D] f32
// =====================================================================
#define LPITCH 72   // LDS row pitch in bf16 (128B data + 16B TDM pad)

__global__ __launch_bounds__(256) void attn_kernel(
    const unsigned short* __restrict__ qh,  // [B,H,S,DK] bf16
    const unsigned short* __restrict__ kh,  // [B,H,S,DK] bf16
    const unsigned short* __restrict__ vT,  // [B,H,DK,S] bf16
    const int* __restrict__ mask,           // [B,S]
    const float* __restrict__ lv,           // [B,H,DK]
    float* __restrict__ concat) {           // [B,S,D]
  __shared__ __align__(16) unsigned short Kt[2][64][LPITCH];  // keys x dk
  __shared__ __align__(16) unsigned short Vt[2][64][LPITCH];  // dk x keys
  __shared__ __align__(16) unsigned short P[8][16][64];       // per-wave C->A staging

  const int b = blockIdx.z, h = blockIdx.y;
  const int tid = threadIdx.x, wave = tid >> 5, lane = tid & 31;
  const int l16 = lane & 15, lh = lane >> 4;
  const int qbase = blockIdx.x * 128 + wave * 16;
  const size_t head = (size_t)(b * HH + h);
  const size_t headQK = head * SS * DKK;

  // Q fragments (16 queries x DK=64 -> two K=32 frags), direct from global
  const unsigned short* qp = qh + headQK + (size_t)(qbase + l16) * DKK;
  v16u qf0 = load_frag(qp, lh);
  v16u qf1 = load_frag(qp + 32, lh);

  v8f acc[4];
#pragma unroll
  for (int dt = 0; dt < 4; ++dt) acc[dt] = (v8f){0.f,0.f,0.f,0.f,0.f,0.f,0.f,0.f};
  float mrow[8], lrow[8];
#pragma unroll
  for (int r = 0; r < 8; ++r) { mrow[r] = -3.0e38f; lrow[r] = 0.f; }
  const float scale = 0.125f;  // 1/sqrt(64)

  auto issue_tile = [&](int kt, int bufi) {
    tdm_load_tile64((unsigned)(uintptr_t)&Kt[bufi][0][0],
                    (const char*)(kh + headQK + (size_t)kt * DKK), DKK);
    tdm_load_tile64((unsigned)(uintptr_t)&Vt[bufi][0][0],
                    (const char*)(vT + head * DKK * SS + kt), SS);
  };
  if (wave == 0) issue_tile(0, 0);

  for (int kt = 0; kt < SS; kt += 64) {
    const int buf = (kt >> 6) & 1;
    if (wave == 0) {
      if (kt + 64 < SS) {
        issue_tile(kt + 64, buf ^ 1);
        __builtin_amdgcn_s_wait_tensorcnt(2);  // in-order: current tile's 2 ops done
      } else {
        __builtin_amdgcn_s_wait_tensorcnt(0);
      }
    }
    __syncthreads();  // staged tile visible to all waves

    // ---- scores: 16 queries x 64 keys (4 sub-tiles of 16 keys) ----
    v8f sc[4];
#pragma unroll
    for (int st = 0; st < 4; ++st) {
      int keyloc = st * 16 + l16;
      const unsigned short* kp = &Kt[buf][keyloc][0];
      v16u bf0 = load_frag(kp, lh);
      v16u bf1 = load_frag(kp + 32, lh);
      v8f c = {0.f,0.f,0.f,0.f,0.f,0.f,0.f,0.f};
      c = wmma_bf16(qf0, bf0, c);
      c = wmma_bf16(qf1, bf1, c);
      bool msk = (mask[(size_t)b * SS + kt + keyloc] == 1);
#pragma unroll
      for (int r = 0; r < 8; ++r) c[r] = msk ? -1e9f : c[r] * scale;
      sc[st] = c;
    }
    // ---- online softmax, per row (half-wave holds one row group) ----
#pragma unroll
    for (int r = 0; r < 8; ++r) {
      float rv = fmaxf(fmaxf(sc[0][r], sc[1][r]), fmaxf(sc[2][r], sc[3][r]));
      for (int o = 8; o > 0; o >>= 1) rv = fmaxf(rv, __shfl_xor(rv, o, 16));
      float mnew = fmaxf(mrow[r], rv);
      float corr = __expf(mrow[r] - mnew);
      mrow[r] = mnew;
      float rsum = 0.f;
#pragma unroll
      for (int st = 0; st < 4; ++st) {
        float p = __expf(sc[st][r] - mnew);
        sc[st][r] = p;
        rsum += p;
      }
      for (int o = 8; o > 0; o >>= 1) rsum += __shfl_xor(rsum, o, 16);
      lrow[r] = lrow[r] * corr + rsum;
#pragma unroll
      for (int st = 0; st < 4; ++st) acc[st][r] *= corr;
    }
    // ---- stage P (C layout) -> LDS, reload as A fragments (wave-local) ----
#pragma unroll
    for (int st = 0; st < 4; ++st)
#pragma unroll
      for (int r = 0; r < 8; ++r)
        P[wave][r + 8 * lh][st * 16 + l16] = f2bf(sc[st][r]);
    const unsigned short* pw = &P[wave][l16][0];
    v16u pf0 = load_frag(pw, lh);
    v16u pf1 = load_frag(pw + 32, lh);
    // ---- acc += P @ V ----
#pragma unroll
    for (int dt = 0; dt < 4; ++dt) {
      const unsigned short* vp = &Vt[buf][dt * 16 + l16][0];
      v16u vf0 = load_frag(vp, lh);
      v16u vf1 = load_frag(vp + 32, lh);
      acc[dt] = wmma_bf16(pf0, vf0, acc[dt]);
      acc[dt] = wmma_bf16(pf1, vf1, acc[dt]);
    }
    __syncthreads();  // all reads done before this buffer is overwritten
  }

  // ---- epilogue: 0.5*acc/l + 0.5*lv -> concat f32 ----
#pragma unroll
  for (int dt = 0; dt < 4; ++dt) {
    int dk = dt * 16 + l16;
    float lvv = lv[head * DKK + dk];
#pragma unroll
    for (int r = 0; r < 8; ++r) {
      int q = qbase + r + 8 * lh;
      float val = 0.5f * acc[dt][r] / lrow[r] + 0.5f * lvv;
      concat[((size_t)b * SS + q) * DD + h * DKK + dk] = val;
    }
  }
}

// =====================================================================
// launcher
// =====================================================================
extern "C" void kernel_launch(void* const* d_in, const int* in_sizes, int n_in,
                              void* d_out, int out_size, void* d_ws, size_t ws_size,
                              hipStream_t stream) {
  const float* q    = (const float*)d_in[0];
  const float* k    = (const float*)d_in[1];
  const float* v    = (const float*)d_in[2];
  const int*   mask = (const int*)  d_in[3];
  const float* Lex  = (const float*)d_in[4];
  const float* Wq   = (const float*)d_in[5];  const float* bq = (const float*)d_in[6];
  const float* Wk   = (const float*)d_in[7];  const float* bk = (const float*)d_in[8];
  const float* Wv   = (const float*)d_in[9];  const float* bv = (const float*)d_in[10];
  const float* Wo   = (const float*)d_in[11]; const float* bo = (const float*)d_in[12];
  const float* Wl   = (const float*)d_in[13]; const float* bl = (const float*)d_in[14];
  float* out = (float*)d_out;

  char* ws = (char*)d_ws;
  size_t off = 0;
  auto alloc = [&](size_t bytes) -> void* {
    void* p = ws + off;
    off += (bytes + 255) & ~(size_t)255;
    return p;
  };
  const size_t headElems = (size_t)BB * HH * SS * DKK;  // 4M
  unsigned short* qhW = (unsigned short*)alloc(headElems * 2);
  unsigned short* khW = (unsigned short*)alloc(headElems * 2);
  unsigned short* vTW = (unsigned short*)alloc(headElems * 2);
  float* concatW  = (float*)alloc((size_t)MM * NN * 4);
  float* lexrowW  = (float*)alloc((size_t)BB * SS * 4);
  float* lexattnW = (float*)alloc((size_t)BB * SS * 4);
  float* lvW      = (float*)alloc((size_t)BB * HH * DKK * 4);

  dim3 gG(NN / 64, MM / 64);  // (8, 128)
  // Q/K/V projections (WMMA bf16)
  gemm_bf16_kernel<1><<<gG, 256, 0, stream>>>(q, Wq, bq, nullptr, qhW);
  gemm_bf16_kernel<1><<<gG, 256, 0, stream>>>(k, Wk, bk, nullptr, khW);
  gemm_bf16_kernel<2><<<gG, 256, 0, stream>>>(v, Wv, bv, nullptr, vTW);

  // lexicon bias path
  lex_matvec_kernel<<<dim3(SS / 256, BB), 256, 0, stream>>>(Lex, Wl, bl, lexrowW);
  lex_softmax_kernel<<<BB, 256, 0, stream>>>(lexrowW, mask, lexattnW);
  lv_kernel<<<BB * HH, 64, 0, stream>>>(lexattnW, vTW, lvW);

  // flash attention (TDM-staged K/V)
  attn_kernel<<<dim3(SS / 128, HH, BB), 256, 0, stream>>>(qhW, khW, vTW, mask, lvW, concatW);

  // output projection
  gemm_bf16_kernel<0><<<gG, 256, 0, stream>>>(concatW, Wo, bo, out, nullptr);
}
